// VectorQuantize_1726576854533
// MI455X (gfx1250) — compile-verified, matrix operand order checked by reading, not run on hardware
//
#include <hip/hip_runtime.h>
#include <hip/hip_bf16.h>
#include <stdint.h>

// ---------------------------------------------------------------------------
// VQ-VAE vector quantize for MI455X (gfx1250, wave32, WMMA).
// Outputs concatenated in d_out (float):
//   quantize_st [N,512], ind [N] (as float), commit_loss [1],
//   new_embed [K,512], new_cluster_size [K], new_embed_avg [K,512]
// ---------------------------------------------------------------------------

#define D        512
#define KCODES   8192
#define NROWS    16384
#define DECAYF   0.8f
#define ONEMDEC  0.2f
#define EPSF     1e-5f
#define NELEMS   8388608.0f

// output offsets (floats)
#define QOFF     0
#define INDOFF   8388608
#define LOSSOFF  8404992
#define NEOFF    8404993
#define NCSOFF   12599297
#define NEAOFF   12607489

// workspace layout (bytes)
#define WS_TOTAL_OFF   0           // 1 float: total cluster mass
#define WS_E2_OFF      256         // 8192 floats: ||embed_k||^2
#define WS_EBLO_OFF    65536       // plane 0: low 16B of each packed B fragment
#define PLANE_BYTES    4194304     // 512 tiles * 16 ksteps * 32 lanes * 16 B
#define WS_EBHI_OFF    (WS_EBLO_OFF + PLANE_BYTES)
#define WS_NEEDED      (WS_EBHI_OFF + PLANE_BYTES)

typedef __attribute__((ext_vector_type(16))) __bf16 v16bf;
typedef __attribute__((ext_vector_type(8)))  float  v8f;

union V16 { v16bf v; unsigned u[8]; uint4 q[2]; };

// pack two f32 -> two bf16 (truncate) in ONE v_perm_b32
static __device__ __forceinline__ unsigned pack2(float hi, float lo) {
    return __builtin_amdgcn_perm(__builtin_bit_cast(unsigned, hi),
                                 __builtin_bit_cast(unsigned, lo), 0x07060302u);
}

// load a 16-float WMMA half-K slab for (row-ptr, k0, half) and pack to v16bf
static __device__ __forceinline__ void load_frag16(const float* __restrict__ r,
                                                   int k0, int half,
                                                   float buf[16]) {
    *(float4*)(buf + 0)  = *(const float4*)(r + k0 + half * 8);
    *(float4*)(buf + 4)  = *(const float4*)(r + k0 + half * 8 + 4);
    *(float4*)(buf + 8)  = *(const float4*)(r + k0 + 16 + half * 8);
    *(float4*)(buf + 12) = *(const float4*)(r + k0 + 16 + half * 8 + 4);
}

static __device__ __forceinline__ v16bf pack16(const float buf[16]) {
    V16 t;
    #pragma unroll
    for (int j = 0; j < 8; ++j) t.u[j] = pack2(buf[2 * j + 1], buf[2 * j]);
    return t.v;
}

// ---------------------------------------------------------------------------
__global__ __launch_bounds__(256) void vq_init(const float* __restrict__ cluster_size,
                                               const float* __restrict__ embed_avg,
                                               float* __restrict__ out) {
    int i = blockIdx.x * 256 + threadIdx.x;
    if (i < KCODES * D) out[NEAOFF + i] = DECAYF * embed_avg[i];
    if (i < KCODES)     out[NCSOFF + i] = DECAYF * cluster_size[i];
    if (i == 0)         out[LOSSOFF]    = 0.0f;
}

// Pre-pack embed into WMMA B-fragment order (bf16), two 16B planes per fragment.
__global__ __launch_bounds__(256) void vq_pack(const float* __restrict__ embed,
                                               unsigned char* __restrict__ ws) {
    int t = blockIdx.x * 256 + threadIdx.x;   // 0 .. 262143 lane-fragments
    int lane = t & 31, ks = (t >> 5) & 15, nt = t >> 9;
    int half = lane >> 4;
    const float* er = embed + (size_t)(nt * 16 + (lane & 15)) * D;
    float buf[16];
    load_frag16(er, ks * 32, half, buf);
    V16 v;
    #pragma unroll
    for (int j = 0; j < 8; ++j) v.u[j] = pack2(buf[2 * j + 1], buf[2 * j]);
    size_t off = (size_t)t * 16;
    *(uint4*)(ws + WS_EBLO_OFF + off) = v.q[0];
    *(uint4*)(ws + WS_EBHI_OFF + off) = v.q[1];
}

// ||embed_k||^2 in full f32
__global__ __launch_bounds__(256) void vq_e2(const float* __restrict__ embed,
                                             float* __restrict__ e2out) {
    int k = blockIdx.x * 256 + threadIdx.x;
    if (k >= KCODES) return;
    const float* er = embed + (size_t)k * D;
    float s = 0.0f;
    for (int j = 0; j < D; j += 4) {
        float4 f = *(const float4*)(er + j);
        s += f.x * f.x + f.y * f.y + f.z * f.z + f.w * f.w;
    }
    e2out[k] = s;
}

// ---------------------------------------------------------------------------
// Shared epilogue: argmin reduce, gather, loss, EMA scatters
// ---------------------------------------------------------------------------
static __device__ __forceinline__ void vq_epilogue(int lane, int m0,
                                                   float minv[8], int mini[8],
                                                   const float* __restrict__ x,
                                                   const float* __restrict__ embed,
                                                   float* __restrict__ out) {
    #pragma unroll
    for (int i = 0; i < 8; ++i) {
        float v = minv[i]; int ix = mini[i];
        #pragma unroll
        for (int off = 8; off >= 1; off >>= 1) {
            float v2 = __shfl_xor(v, off);
            int   i2 = __shfl_xor(ix, off);
            if (v2 < v || (v2 == v && i2 < ix)) { v = v2; ix = i2; }
        }
        mini[i] = ix;
    }
    int idxAll[16];
    #pragma unroll
    for (int i = 0; i < 8; ++i) {
        idxAll[i]     = __shfl(mini[i], 0);
        idxAll[8 + i] = __shfl(mini[i], 16);
    }
    if (lane < 16) out[INDOFF + m0 + lane] = (float)idxAll[lane];
    if (lane == 0) {
        #pragma unroll
        for (int r = 0; r < 16; ++r)
            atomicAdd(&out[NCSOFF + idxAll[r]], ONEMDEC);
    }
    float lossp = 0.0f;
    for (int r = 0; r < 16; ++r) {
        const int idx = idxAll[r];
        const float* er = embed + (size_t)idx * D;
        const float* xr = x + (size_t)(m0 + r) * D;
        float* q  = out + QOFF + (size_t)(m0 + r) * D;
        float* ea = out + NEAOFF + (size_t)idx * D;
        for (int j = lane * 4; j < D; j += 128) {
            float4 ev = *(const float4*)(er + j);
            float4 xv = *(const float4*)(xr + j);
            *(float4*)(q + j) = ev;
            float d0 = ev.x - xv.x, d1 = ev.y - xv.y, d2 = ev.z - xv.z, d3 = ev.w - xv.w;
            lossp += d0 * d0 + d1 * d1 + d2 * d2 + d3 * d3;
            atomicAdd(ea + j + 0, ONEMDEC * xv.x);
            atomicAdd(ea + j + 1, ONEMDEC * xv.y);
            atomicAdd(ea + j + 2, ONEMDEC * xv.z);
            atomicAdd(ea + j + 3, ONEMDEC * xv.w);
        }
    }
    #pragma unroll
    for (int off = 16; off >= 1; off >>= 1) lossp += __shfl_xor(lossp, off);
    if (lane == 0) atomicAdd(&out[LOSSOFF], lossp * (1.0f / NELEMS));
}

// ---------------------------------------------------------------------------
// FAST path: B fragments pre-packed bf16 in ws, e2 precomputed.
// Hot loop per K-step: 2x global_load_b128 + 1x v_wmma_f32_16x16x32_bf16.
// ---------------------------------------------------------------------------
__global__ __launch_bounds__(256) void vq_main_fast(const float* __restrict__ x,
                                                    const float* __restrict__ embed,
                                                    const unsigned char* __restrict__ ws,
                                                    float* __restrict__ out) {
    const int lane = threadIdx.x & 31;
    const int wid  = (blockIdx.x * blockDim.x + threadIdx.x) >> 5;
    const int m0   = wid * 16;
    const int half = lane >> 4;
    const int row  = lane & 15;
    const float* __restrict__ e2ws = (const float*)(ws + WS_E2_OFF);
    const unsigned char* __restrict__ eblo = ws + WS_EBLO_OFF + (size_t)lane * 16;
    const unsigned char* __restrict__ ebhi = ws + WS_EBHI_OFF + (size_t)lane * 16;

    // A fragments (16 rows x 512 d, bf16) resident in VGPRs; x^2 on the side
    v16bf a[16];
    float x2p = 0.0f;
    {
        const float* xr = x + (size_t)(m0 + row) * D;
        #pragma unroll
        for (int ks = 0; ks < 16; ++ks) {
            float buf[16];
            load_frag16(xr, ks * 32, half, buf);
            #pragma unroll
            for (int j = 0; j < 16; ++j) x2p += buf[j] * buf[j];
            a[ks] = pack16(buf);
        }
    }
    float x2full = x2p + __shfl_xor(x2p, 16);
    float x2r[8];
    #pragma unroll
    for (int i = 0; i < 8; ++i) x2r[i] = __shfl(x2full, half * 8 + i);

    float minv[8];
    int   mini[8];
    #pragma unroll
    for (int i = 0; i < 8; ++i) { minv[i] = 3.4e38f; mini[i] = 0; }

    for (int nt = 0; nt < KCODES / 16; ++nt) {
        const size_t tb = (size_t)nt * 8192;             // 16 ksteps * 512 B
        __builtin_prefetch(eblo + tb + 8192, 0, 1);      // next tile
        __builtin_prefetch(ebhi + tb + 8192, 0, 1);
        v8f c0 = {0.f, 0.f, 0.f, 0.f, 0.f, 0.f, 0.f, 0.f};
        v8f c1 = {0.f, 0.f, 0.f, 0.f, 0.f, 0.f, 0.f, 0.f};
        #pragma unroll
        for (int ks = 0; ks < 16; ks += 2) {
            V16 b0, b1;
            b0.q[0] = *(const uint4*)(eblo + tb + (size_t)ks * 512);
            b0.q[1] = *(const uint4*)(ebhi + tb + (size_t)ks * 512);
            b1.q[0] = *(const uint4*)(eblo + tb + (size_t)(ks + 1) * 512);
            b1.q[1] = *(const uint4*)(ebhi + tb + (size_t)(ks + 1) * 512);
            c0 = __builtin_amdgcn_wmma_f32_16x16x32_bf16(
                     false, a[ks], false, b0.v, (short)0, c0, false, false);
            c1 = __builtin_amdgcn_wmma_f32_16x16x32_bf16(
                     false, a[ks + 1], false, b1.v, (short)0, c1, false, false);
        }
        const float e2  = e2ws[nt * 16 + row];
        const int   myn = nt * 16 + row;
        #pragma unroll
        for (int i = 0; i < 8; ++i) {
            float d2 = fmaxf(fmaf(-2.0f, c0[i] + c1[i], x2r[i] + e2), 0.0f);
            if (d2 < minv[i]) { minv[i] = d2; mini[i] = myn; }
        }
    }
    vq_epilogue(lane, m0, minv, mini, x, embed, out);
}

// ---------------------------------------------------------------------------
// FALLBACK path (small ws): inline conversion with v_perm packing.
// ---------------------------------------------------------------------------
__global__ __launch_bounds__(256) void vq_main_slow(const float* __restrict__ x,
                                                    const float* __restrict__ embed,
                                                    float* __restrict__ out) {
    const int lane = threadIdx.x & 31;
    const int wid  = (blockIdx.x * blockDim.x + threadIdx.x) >> 5;
    const int m0   = wid * 16;
    const int half = lane >> 4;
    const int row  = lane & 15;

    v16bf a[16];
    float x2p = 0.0f;
    {
        const float* xr = x + (size_t)(m0 + row) * D;
        #pragma unroll
        for (int ks = 0; ks < 16; ++ks) {
            float buf[16];
            load_frag16(xr, ks * 32, half, buf);
            #pragma unroll
            for (int j = 0; j < 16; ++j) x2p += buf[j] * buf[j];
            a[ks] = pack16(buf);
        }
    }
    float x2full = x2p + __shfl_xor(x2p, 16);
    float x2r[8];
    #pragma unroll
    for (int i = 0; i < 8; ++i) x2r[i] = __shfl(x2full, half * 8 + i);

    float minv[8];
    int   mini[8];
    #pragma unroll
    for (int i = 0; i < 8; ++i) { minv[i] = 3.4e38f; mini[i] = 0; }

    for (int nt = 0; nt < KCODES / 16; ++nt) {
        const float* er = embed + (size_t)(nt * 16 + row) * D;
        v8f c = {0.f, 0.f, 0.f, 0.f, 0.f, 0.f, 0.f, 0.f};
        float e2p = 0.0f;
        #pragma unroll
        for (int ks = 0; ks < 16; ++ks) {
            float buf[16];
            load_frag16(er, ks * 32, half, buf);
            #pragma unroll
            for (int j = 0; j < 16; ++j) e2p += buf[j] * buf[j];
            v16bf b = pack16(buf);
            c = __builtin_amdgcn_wmma_f32_16x16x32_bf16(
                    false, a[ks], false, b, (short)0, c, false, false);
        }
        const float e2  = e2p + __shfl_xor(e2p, 16);
        const int   myn = nt * 16 + row;
        #pragma unroll
        for (int i = 0; i < 8; ++i) {
            float d2 = fmaxf(fmaf(-2.0f, c[i], x2r[i] + e2), 0.0f);
            if (d2 < minv[i]) { minv[i] = d2; mini[i] = myn; }
        }
    }
    vq_epilogue(lane, m0, minv, mini, x, embed, out);
}

// ---------------------------------------------------------------------------
__global__ __launch_bounds__(256) void vq_total(const float* __restrict__ out,
                                                float* __restrict__ ws) {
    __shared__ float red[256];
    float s = 0.0f;
    for (int i = threadIdx.x; i < KCODES; i += 256) s += out[NCSOFF + i];
    red[threadIdx.x] = s;
    __syncthreads();
    for (int o = 128; o > 0; o >>= 1) {
        if (threadIdx.x < (unsigned)o) red[threadIdx.x] += red[threadIdx.x + o];
        __syncthreads();
    }
    if (threadIdx.x == 0) ws[0] = red[0];
}

__global__ __launch_bounds__(256) void vq_newembed(float* __restrict__ out,
                                                   const float* __restrict__ ws) {
    int i = blockIdx.x * 256 + threadIdx.x;
    if (i >= KCODES * D) return;
    int k = i >> 9;
    float total = ws[0];
    float cs = out[NCSOFF + k];
    float smoothed = (cs + EPSF) / (total + KCODES * EPSF) * total;
    out[NEOFF + i] = out[NEAOFF + i] / smoothed;
}

// ---------------------------------------------------------------------------
extern "C" void kernel_launch(void* const* d_in, const int* in_sizes, int n_in,
                              void* d_out, int out_size, void* d_ws, size_t ws_size,
                              hipStream_t stream) {
    const float* x            = (const float*)d_in[0];
    const float* embed        = (const float*)d_in[1];
    const float* cluster_size = (const float*)d_in[2];
    const float* embed_avg    = (const float*)d_in[3];
    float* out = (float*)d_out;
    unsigned char* ws = (unsigned char*)d_ws;

    vq_init<<<(KCODES * D + 255) / 256, 256, 0, stream>>>(cluster_size, embed_avg, out);

    if (ws_size >= (size_t)WS_NEEDED) {
        vq_pack<<<(KCODES / 16) * 16 * 32 / 256, 256, 0, stream>>>(embed, ws);
        vq_e2<<<KCODES / 256, 256, 0, stream>>>(embed, (float*)(ws + WS_E2_OFF));
        vq_main_fast<<<NROWS / (16 * 8), 256, 0, stream>>>(x, embed, ws, out);
    } else {
        vq_main_slow<<<NROWS / (16 * 8), 256, 0, stream>>>(x, embed, out);
    }
    vq_total<<<1, 256, 0, stream>>>(out, (float*)(ws + WS_TOTAL_OFF));
    vq_newembed<<<(KCODES * D + 255) / 256, 256, 0, stream>>>(out, (float*)(ws + WS_TOTAL_OFF));
}
